// InstanceSelectorMIL_15006615733114
// MI455X (gfx1250) — matrix operand order repaired and checked by reading, not compiled, not run on hardware
//
#include <hip/hip_runtime.h>
#include <hip/hip_bf16.h>

typedef __attribute__((ext_vector_type(16))) __bf16 v16bf;
typedef __attribute__((ext_vector_type(8)))  __bf16 v8bf;
typedef __attribute__((ext_vector_type(8)))  float  v8f;
typedef __attribute__((ext_vector_type(4)))  float  v4f;

#define DD   512
#define HH   256
#define NB   32
#define NI   8192
#define TOPK 10
#define NEG_INF (-3.402823466e+38f)

#define WAVES      16            // waves per block in the GEMM kernel
#define LDS_STRIDE 520           // 512 + 8 bf16 pad -> conflict-free b128 reads
#define LDS_BYTES  (HH * LDS_STRIDE * 2)   // 266240 B <= 320 KB WGP LDS

// ---------------------------------------------------------------------------
// Kernel 0: W1 [512,256] fp32 (row-major d x h) -> W1t [256][512] bf16
// (column-major so WMMA B fragments are contiguous 16B loads)
// ---------------------------------------------------------------------------
__global__ __launch_bounds__(256) void w1_transpose_bf16(
    const float* __restrict__ W1, __bf16* __restrict__ W1t) {
  int i = blockIdx.x * 256 + threadIdx.x;   // 0 .. 131071
  int k = i >> 8;                           // 0..511
  int c = i & 255;                          // 0..255
  W1t[(size_t)c * DD + k] = (__bf16)W1[(size_t)k * HH + c];
}

// ---------------------------------------------------------------------------
// Kernel 1: scores[row] = relu(X[row]·W1 + b1) · W2 + b2
// 16 waves/block share a 260 KB LDS copy of W1t; one wave per 16-row tile.
// bf16 WMMA (16x16x32), fp32 accumulate. A stream software-pipelined from
// HBM (nontemporal); B fragments 4-deep rotated from LDS so every WMMA has
// >= 3 newer DS loads outstanding (no s_wait_dscnt 0 stalls).
// ---------------------------------------------------------------------------
__global__ __launch_bounds__(512) void selector_scores(
    const float* __restrict__ X,        // [NB*NI, DD]
    const __bf16* __restrict__ W1t,     // [HH][DD] bf16 (transposed)
    const float* __restrict__ b1,       // [HH]
    const float* __restrict__ W2,       // [HH]
    const float* __restrict__ b2,       // [1]
    float* __restrict__ scores)         // [NB*NI]
{
  extern __shared__ char smem_raw[];
  __bf16* w1s = (__bf16*)smem_raw;      // [HH][LDS_STRIDE]

  const int tid = threadIdx.x;

  // --- cooperative copy: W1t (contiguous [c][k] bf16) -> padded LDS ---
  for (int j = tid; j < (HH * DD) / 8; j += 512) {   // 16384 16B chunks
    const int c  = j >> 6;          // column (0..255)
    const int kc = j & 63;          // 8-elem chunk within column
    v8bf src = *(const v8bf*)(W1t + ((size_t)c << 9) + (kc << 3));
    *(v8bf*)(w1s + c * LDS_STRIDE + (kc << 3)) = src;
  }
  __syncthreads();

  const int lane    = tid & 31;
  const int wave    = tid >> 5;
  const int tile    = blockIdx.x * WAVES + wave;   // 16384 tiles total
  const int rowBase = tile * 16;
  const int m       = lane & 15;    // row within tile (A), col (B/C)
  const int g       = lane >> 4;    // lane-half selector

  const float* arow = X + (size_t)(rowBase + m) * DD;

  v8f zero = {};
  v8f acc[16];
#pragma unroll
  for (int nt = 0; nt < 16; ++nt) acc[nt] = zero;

  // --- prologue: load A floats for kb = 0 (nontemporal: A is streamed once)
  const v4f* p0 = (const v4f*)(arow + g * 8);
  v4f f0 = __builtin_nontemporal_load(p0);
  v4f f1 = __builtin_nontemporal_load(p0 + 1);
  v4f f2 = __builtin_nontemporal_load(p0 + 4);   // +16 floats
  v4f f3 = __builtin_nontemporal_load(p0 + 5);

  for (int kb = 0; kb < 16; ++kb) {            // K = 512, 32 per step
    // --- build A fragment (16x32 bf16). Lane m+16g holds row m,
    //     elements 0..7 = K(kb*32 + 8g + 0..7), elements 8..15 = +16.
    v16bf a;
#pragma unroll
    for (int e = 0; e < 4; ++e) {
      a[e]      = (__bf16)f0[e];
      a[e + 4]  = (__bf16)f1[e];
      a[e + 8]  = (__bf16)f2[e];
      a[e + 12] = (__bf16)f3[e];
    }

    // --- prefetch next K-block's A while the 16 WMMAs run ---
    if (kb < 15) {
      const v4f* pn = (const v4f*)(arow + (kb + 1) * 32 + g * 8);
      f0 = __builtin_nontemporal_load(pn);
      f1 = __builtin_nontemporal_load(pn + 1);
      f2 = __builtin_nontemporal_load(pn + 4);
      f3 = __builtin_nontemporal_load(pn + 5);
    }

    // --- B fragments from LDS (32x16 bf16): lane n+16g holds column n,
    //     elements 0..7 = K(kb*32 + 8g + 0..7), elements 8..15 = +16.
    //     4-deep rotating buffer: loads run 4 N-tiles ahead of use.
    const __bf16* bbase = w1s + kb * 32 + g * 8;

    v16bf B[4];
#pragma unroll
    for (int q = 0; q < 4; ++q) {
      const __bf16* pb = bbase + (size_t)(q * 16 + m) * LDS_STRIDE;
      v8bf lo = *(const v8bf*)pb;
      v8bf hi = *(const v8bf*)(pb + 16);
#pragma unroll
      for (int e = 0; e < 8; ++e) { B[q][e] = lo[e]; B[q][e + 8] = hi[e]; }
    }

#pragma unroll
    for (int nt = 0; nt < 16; ++nt) {
      acc[nt] = __builtin_amdgcn_wmma_f32_16x16x32_bf16(
          false, a, false, B[nt & 3], (short)0, acc[nt], false, false);
      if (nt < 12) {
        const __bf16* pb = bbase + (size_t)((nt + 4) * 16 + m) * LDS_STRIDE;
        v8bf lo = *(const v8bf*)pb;
        v8bf hi = *(const v8bf*)(pb + 16);
#pragma unroll
        for (int e = 0; e < 8; ++e) { B[nt & 3][e] = lo[e]; B[nt & 3][e + 8] = hi[e]; }
      }
    }
  }

  // Epilogue: h = relu(acc + b1), partial score = h * W2, reduce over N.
  // C layout: acc[nt][r] = element (row rowBase + r + 8g, col nt*16 + m).
  float partial[8];
#pragma unroll
  for (int r = 0; r < 8; ++r) partial[r] = 0.0f;
#pragma unroll
  for (int nt = 0; nt < 16; ++nt) {
    const int colg = nt * 16 + m;
    const float bb = b1[colg];
    const float w2 = W2[colg];
#pragma unroll
    for (int r = 0; r < 8; ++r) {
      float h = acc[nt][r] + bb;
      h = h > 0.0f ? h : 0.0f;
      partial[r] += h * w2;
    }
  }
  // Sum across the 16 lanes of each half (each holds distinct N columns).
#pragma unroll
  for (int r = 0; r < 8; ++r) {
    float v = partial[r];
    v += __shfl_xor(v, 1);
    v += __shfl_xor(v, 2);
    v += __shfl_xor(v, 4);
    v += __shfl_xor(v, 8);
    partial[r] = v;
  }
  const float bias2 = b2[0];
  if (m == 0) {                       // lanes 0 (rows 0..7) and 16 (rows 8..15)
    const int rb = rowBase + g * 8;
#pragma unroll
    for (int r = 0; r < 8; ++r) scores[rb + r] = partial[r] + bias2;
  }
}

// ---------------------------------------------------------------------------
// Kernel 2: per-bag top-10 (lowest-index tie-break), softmax, gather+mean,
// classifier. One block (256 thr) per bag.
// ---------------------------------------------------------------------------
__global__ __launch_bounds__(256) void mil_finalize(
    const float* __restrict__ X,        // [NB*NI, DD]
    const float* __restrict__ scores,   // [NB*NI]
    const float* __restrict__ Wc,       // [DD,2]
    const float* __restrict__ bc,       // [2]
    float* __restrict__ out)
{
  __shared__ float sdata[NI];
  __shared__ float red_s[256];
  __shared__ int   red_i[256];
  __shared__ float topk_s[TOPK];
  __shared__ int   topk_i[TOPK];
  __shared__ float bfeat[DD];

  const int bag = blockIdx.x;
  const int tid = threadIdx.x;
  const float* sc = scores + (size_t)bag * NI;

  for (int i = tid; i < NI; i += 256) sdata[i] = sc[i];
  __syncthreads();

  // --- iterative top-10 extraction (descending, first-index tie-break) ---
  for (int t = 0; t < TOPK; ++t) {
    float best = NEG_INF; int bidx = NI;
    for (int i = tid; i < NI; i += 256) {
      float v = sdata[i];
      if (v > best || (v == best && i < bidx)) { best = v; bidx = i; }
    }
    red_s[tid] = best; red_i[tid] = bidx;
    __syncthreads();
    for (int s = 128; s > 0; s >>= 1) {
      if (tid < s) {
        float v = red_s[tid + s]; int j = red_i[tid + s];
        if (v > red_s[tid] || (v == red_s[tid] && j < red_i[tid])) {
          red_s[tid] = v; red_i[tid] = j;
        }
      }
      __syncthreads();
    }
    if (tid == 0) {
      topk_s[t] = red_s[0];
      topk_i[t] = red_i[0];
      sdata[red_i[0]] = NEG_INF;        // remove from future rounds
    }
    __syncthreads();
  }

  // --- softmax over the bag's scores ---
  const float maxv = topk_s[0];
  float lsum = 0.0f;
  for (int i = tid; i < NI; i += 256) {
    float e = __expf(sc[i] - maxv);     // original scores from global
    sdata[i] = e;
    lsum += e;
  }
  red_s[tid] = lsum;
  __syncthreads();
  for (int s = 128; s > 0; s >>= 1) {
    if (tid < s) red_s[tid] += red_s[tid + s];
    __syncthreads();
  }
  const float inv = 1.0f / red_s[0];
  float* att = out + NB * 2 + (size_t)bag * NI;
  for (int i = tid; i < NI; i += 256) att[i] = sdata[i] * inv;

  // --- bag features: mean of the 10 selected instances ---
  float* obf = out + NB * 2 + (size_t)NB * NI + NB * TOPK + (size_t)bag * DD;
  for (int d = tid; d < DD; d += 256) {
    float s = 0.0f;
#pragma unroll
    for (int t = 0; t < TOPK; ++t)
      s += X[((size_t)bag * NI + topk_i[t]) * DD + d];
    float mv = s * (1.0f / TOPK);
    bfeat[d] = mv;
    obf[d] = mv;
  }
  __syncthreads();

  // --- logits = bag_features @ Wc + bc ---
  for (int c = 0; c < 2; ++c) {
    float p = 0.0f;
    for (int d = tid; d < DD; d += 256) p += bfeat[d] * Wc[d * 2 + c];
    red_s[tid] = p;
    __syncthreads();
    for (int s = 128; s > 0; s >>= 1) {
      if (tid < s) red_s[tid] += red_s[tid + s];
      __syncthreads();
    }
    if (tid == 0) out[bag * 2 + c] = red_s[0] + bc[c];
    __syncthreads();
  }

  // --- top-k indices (as float, per harness output dtype) ---
  if (tid < TOPK)
    out[NB * 2 + (size_t)NB * NI + bag * TOPK + tid] = (float)topk_i[tid];
}

// ---------------------------------------------------------------------------
extern "C" void kernel_launch(void* const* d_in, const int* in_sizes, int n_in,
                              void* d_out, int out_size, void* d_ws, size_t ws_size,
                              hipStream_t stream) {
  const float* X  = (const float*)d_in[0];   // [32,8192,512]
  const float* W1 = (const float*)d_in[1];   // [512,256]
  const float* b1 = (const float*)d_in[2];   // [256]
  const float* W2 = (const float*)d_in[3];   // [256,1]
  const float* b2 = (const float*)d_in[4];   // [1]
  const float* Wc = (const float*)d_in[5];   // [512,2]
  const float* bc = (const float*)d_in[6];   // [2]
  float* out = (float*)d_out;

  // workspace: scores fp32 (1 MB) | W1t bf16 (256 KB)
  float*  scores = (float*)d_ws;
  __bf16* W1t    = (__bf16*)((char*)d_ws + (size_t)NB * NI * sizeof(float));

  // allow 260 KB dynamic LDS (CDNA5 WGP supports 320 KB); non-stream API,
  // capture-safe and idempotent.
  (void)hipFuncSetAttribute((const void*)selector_scores,
                            hipFuncAttributeMaxDynamicSharedMemorySize,
                            LDS_BYTES);

  w1_transpose_bf16<<<(DD * HH) / 256, 256, 0, stream>>>(W1, W1t);
  selector_scores<<<(NB * NI / 16) / WAVES, 512, LDS_BYTES, stream>>>(
      X, W1t, b1, W2, b2, scores);
  mil_finalize<<<NB, 256, 0, stream>>>(X, scores, Wc, bc, out);
}
